// DecoderLayer_38757784879515
// MI455X (gfx1250) — compile-verified, matrix operand order checked by reading, not run on hardware
//
#include <hip/hip_runtime.h>
#include <hip/hip_bf16.h>

// ---------------- problem constants ----------------
#define B_    4
#define S_    512
#define D_    512
#define H_    8
#define HS_   64
#define LAT_  64
#define E_    16
#define HID_  1365
#define HIDP_ 1376          // HID padded to multiple of 32
#define TOPK_ 4
#define T_    (B_*S_)       // 2048 tokens
#define EPS_  1.1920929e-07f

typedef __attribute__((ext_vector_type(16))) __bf16 bf16x16;
typedef __attribute__((ext_vector_type(8)))  __bf16 bf16x8;
typedef __attribute__((ext_vector_type(8)))  float  f32x8;
typedef __attribute__((ext_vector_type(4)))  float  f32x4;

// gemm mode flags
#define F_EXPERT  1   // rows & row-base come from expOff[z]
#define F_AGATHER 2   // A rows gathered through slotTok
#define F_SCATTER 4   // epilogue: atomicAdd into C[token], scaled by slotW
#define F_ADD     8   // epilogue: += aux[row*ldc+n]  (residual)
#define F_BIAS    16  // epilogue: += aux[n]

struct GemmArgs {
  const float* A; const float* W; float* C; const float* aux;
  const int* expOff; const int* slotTok; const float* slotW;
  long long sAm, sAk, sWn, sWk, ldc;
  long long bA1, bA2, bW1, bW2, bC1, bC2;   // batch strides: z = z1*nb2 + z2
  int nb2, M, N, K, flags;
  float alpha;
};

__device__ inline __bf16 f2bf(float f) { return (__bf16)f; }   // v_cvt, RNE

// ---------------------------------------------------------------------------
// Generic bf16-WMMA GEMM: C[m,n] (+)= alpha * sum_k A[m,k] * W[n,k]
// Block tile 128x64, 128 threads = 4 waves; each wave owns a 32x64 strip:
// 2 M-subtiles x 4 N-subtiles = 8x v_wmma_f32_16x16x32_bf16 per K-step.
// FAST=true: unit K strides, 16B-aligned rows, K%32==0. Row pointers (incl.
// the token gather) are hoisted out of the K loop; OOB rows are clamped and
// zeroed with a select -> branch-free staging, b128 loads, b128 LDS stores.
// ---------------------------------------------------------------------------
template <bool FAST>
__global__ __launch_bounds__(128) void k_gemm(GemmArgs p) {
  const int tid  = threadIdx.x;
  const int lane = tid & 31;
  const int wv   = tid >> 5;
  const int z    = blockIdx.z;
  const int z1   = z / p.nb2;
  const int z2   = z - z1 * p.nb2;

  const long long offA = (long long)z1 * p.bA1 + (long long)z2 * p.bA2;
  const long long offW = (long long)z1 * p.bW1 + (long long)z2 * p.bW2;
  const long long offC = (long long)z1 * p.bC1 + (long long)z2 * p.bC2;

  int rowBase = 0;
  int rows    = p.M;
  if (p.flags & F_EXPERT) {
    rowBase = p.expOff[z];
    rows    = p.expOff[z + 1] - rowBase;
  }

  const int m0 = blockIdx.y * 128;
  const int n0 = blockIdx.x * 64;
  if (m0 >= rows) return;

  __shared__ __align__(16) __bf16 As[128 * 40];  // 40-elem rows: 80B, 16B-aligned chunks
  __shared__ __align__(16) __bf16 Bs[64 * 40];

  f32x8 acc[2][4];
  #pragma unroll
  for (int a = 0; a < 2; ++a)
    #pragma unroll
    for (int i = 0; i < 4; ++i)
      #pragma unroll
      for (int j = 0; j < 8; ++j) acc[a][i][j] = 0.0f;

  const int hf  = lane >> 4;
  const int l16 = lane & 15;

  // ---- K-invariant staging state (FAST): row ptrs + validity, gather once ----
  const float* aPtr[4]; bool aOk[4]; int aR[4], aCo[4];
  const float* wPtr[2]; bool wOk[2]; int wR[2], wCo[2];
  if (FAST) {
    #pragma unroll
    for (int it = 0; it < 4; ++it) {
      int g = tid + it * 128;
      aR[it]  = g >> 2;               // row within 128-row tile
      aCo[it] = (g & 3) * 8;          // column octet
      int gm  = m0 + aR[it];
      aOk[it] = gm < rows;
      int gmc = aOk[it] ? gm : (rows - 1);
      long long ar = (p.flags & F_AGATHER) ? (long long)p.slotTok[rowBase + gmc]
                                           : (long long)(rowBase + gmc);
      aPtr[it] = p.A + offA + ar * p.sAm + aCo[it];
    }
    #pragma unroll
    for (int it = 0; it < 2; ++it) {
      int g = tid + it * 128;
      wR[it]  = g >> 2;
      wCo[it] = (g & 3) * 8;
      int gn  = n0 + wR[it];
      wOk[it] = gn < p.N;
      int gnc = wOk[it] ? gn : (p.N - 1);
      wPtr[it] = p.W + offW + (long long)gnc * p.sWn + wCo[it];
    }
  }

  for (int k0 = 0; k0 < p.K; k0 += 32) {
    if (FAST) {
      // ---- A: 128x32 tile, one column-octet (2x b128 load, 1x b128 store) ----
      #pragma unroll
      for (int it = 0; it < 4; ++it) {
        f32x4 v0 = *(const f32x4*)(aPtr[it] + k0);
        f32x4 v1 = *(const f32x4*)(aPtr[it] + k0 + 4);
        if (it == 0 && k0 + 32 < p.K) __builtin_prefetch(aPtr[it] + k0 + 32, 0, 3);
        if (!aOk[it]) {
          #pragma unroll
          for (int q = 0; q < 4; ++q) { v0[q] = 0.0f; v1[q] = 0.0f; }
        }
        bf16x8 hv;
        #pragma unroll
        for (int q = 0; q < 4; ++q) { hv[q] = f2bf(v0[q]); hv[q + 4] = f2bf(v1[q]); }
        *(bf16x8*)(&As[aR[it] * 40 + aCo[it]]) = hv;
      }
      // ---- W: 64x32 tile ----
      #pragma unroll
      for (int it = 0; it < 2; ++it) {
        f32x4 v0 = *(const f32x4*)(wPtr[it] + k0);
        f32x4 v1 = *(const f32x4*)(wPtr[it] + k0 + 4);
        if (it == 0 && k0 + 32 < p.K) __builtin_prefetch(wPtr[it] + k0 + 32, 0, 3);
        if (!wOk[it]) {
          #pragma unroll
          for (int q = 0; q < 4; ++q) { v0[q] = 0.0f; v1[q] = 0.0f; }
        }
        bf16x8 hv;
        #pragma unroll
        for (int q = 0; q < 4; ++q) { hv[q] = f2bf(v0[q]); hv[q + 4] = f2bf(v1[q]); }
        *(bf16x8*)(&Bs[wR[it] * 40 + wCo[it]]) = hv;
      }
    } else {
      // ---- general strided path (small GEMMs only) ----
      #pragma unroll
      for (int it = 0; it < 8; ++it) {
        int g = tid + it * 128;
        int r = g >> 3, cq = (g & 7) * 4;
        int gm = m0 + r, gk = k0 + cq;
        f32x4 v; v[0] = v[1] = v[2] = v[3] = 0.0f;
        if (gm < rows) {
          long long ar = (p.flags & F_AGATHER) ? (long long)p.slotTok[rowBase + gm]
                                               : (long long)(rowBase + gm);
          const float* src = p.A + offA + ar * p.sAm + (long long)gk * p.sAk;
          #pragma unroll
          for (int q = 0; q < 4; ++q)
            if (gk + q < p.K) v[q] = src[(long long)q * p.sAk];
        }
        As[r * 40 + cq + 0] = f2bf(v[0]); As[r * 40 + cq + 1] = f2bf(v[1]);
        As[r * 40 + cq + 2] = f2bf(v[2]); As[r * 40 + cq + 3] = f2bf(v[3]);
      }
      #pragma unroll
      for (int it = 0; it < 4; ++it) {
        int g = tid + it * 128;
        int r = g >> 3, cq = (g & 7) * 4;
        int gn = n0 + r, gk = k0 + cq;
        f32x4 v; v[0] = v[1] = v[2] = v[3] = 0.0f;
        if (gn < p.N) {
          const float* src = p.W + offW + (long long)gn * p.sWn + (long long)gk * p.sWk;
          #pragma unroll
          for (int q = 0; q < 4; ++q)
            if (gk + q < p.K) v[q] = src[(long long)q * p.sWk];
        }
        Bs[r * 40 + cq + 0] = f2bf(v[0]); Bs[r * 40 + cq + 1] = f2bf(v[1]);
        Bs[r * 40 + cq + 2] = f2bf(v[2]); Bs[r * 40 + cq + 3] = f2bf(v[3]);
      }
    }
    __syncthreads();

    // ---- fragments + 8 WMMA per wave ----
    bf16x16 af[2];
    #pragma unroll
    for (int ms = 0; ms < 2; ++ms) {
      const __bf16* ap = &As[(wv * 32 + ms * 16 + l16) * 40 + hf * 8];
      bf16x8 alo = *(const bf16x8*)ap;
      bf16x8 ahi = *(const bf16x8*)(ap + 16);
      af[ms] = __builtin_shufflevector(alo, ahi,
          0,1,2,3,4,5,6,7,8,9,10,11,12,13,14,15);
    }
    #pragma unroll
    for (int tn = 0; tn < 4; ++tn) {
      const __bf16* bp = &Bs[(tn * 16 + l16) * 40 + hf * 8];
      bf16x8 blo = *(const bf16x8*)bp;
      bf16x8 bhi = *(const bf16x8*)(bp + 16);
      bf16x16 bfr = __builtin_shufflevector(blo, bhi,
          0,1,2,3,4,5,6,7,8,9,10,11,12,13,14,15);
      #pragma unroll
      for (int ms = 0; ms < 2; ++ms)
        acc[ms][tn] = __builtin_amdgcn_wmma_f32_16x16x32_bf16(
            false, af[ms], false, bfr, (short)0, acc[ms][tn], false, false);
    }
    __syncthreads();
  }

  // ---- epilogue: C/D layout: lane%16 = col, VGPR e + 8*(lane/16) = row ----
  #pragma unroll
  for (int ms = 0; ms < 2; ++ms) {
    #pragma unroll
    for (int tn = 0; tn < 4; ++tn) {
      int n = n0 + tn * 16 + l16;
      if (n >= p.N) continue;
      #pragma unroll
      for (int e = 0; e < 8; ++e) {
        int m = m0 + wv * 32 + ms * 16 + hf * 8 + e;
        if (m >= rows) continue;
        float val = acc[ms][tn][e] * p.alpha;
        if (p.flags & F_SCATTER) {
          long long trow = (p.flags & F_EXPERT) ? (long long)p.slotTok[rowBase + m]
                                                : (long long)m;
          float wgt = p.slotW ? p.slotW[rowBase + m] : 1.0f;
          atomicAdd(p.C + offC + trow * p.ldc + n, wgt * val);
        } else {
          long long crow = (p.flags & F_EXPERT) ? (long long)(rowBase + m)
                                                : (long long)m;
          if (p.flags & F_BIAS) val += p.aux[n];
          if (p.flags & F_ADD)  val += p.aux[crow * p.ldc + n];
          p.C[offC + crow * p.ldc + n] = val;
        }
      }
    }
  }
}

// ---------------- elementwise / reduction kernels ----------------
__device__ inline float wredSum(float v) {
  #pragma unroll
  for (int o = 16; o > 0; o >>= 1) v += __shfl_xor(v, o, 32);
  return v;
}
__device__ inline float wredMax(float v) {
  #pragma unroll
  for (int o = 16; o > 0; o >>= 1) v = fmaxf(v, __shfl_xor(v, o, 32));
  return v;
}

__global__ __launch_bounds__(256) void k_rmsnorm(const float* __restrict__ x,
                                                 const float* __restrict__ g,
                                                 float* __restrict__ o, int D) {
  long long row = blockIdx.x;
  const float* xr = x + row * D;
  float* orow = o + row * D;
  int lane = threadIdx.x & 31, wv = threadIdx.x >> 5;
  __shared__ float sh[8];
  float ss = 0.f;
  for (int i = threadIdx.x; i < D; i += 256) { float v = xr[i]; ss += v * v; }
  ss = wredSum(ss);
  if (lane == 0) sh[wv] = ss;
  __syncthreads();
  float tot = 0.f;
  #pragma unroll
  for (int i = 0; i < 8; ++i) tot += sh[i];
  float r = rsqrtf(tot / (float)D + EPS_);
  for (int i = threadIdx.x; i < D; i += 256) orow[i] = xr[i] * r * g[i];
}

__global__ __launch_bounds__(256) void k_softmax_causal(float* sc, int S) {
  long long r = blockIdx.x;              // r = (b*H+h)*S + s
  int s = (int)(r % S);
  float* row = sc + r * (long long)S;
  int lane = threadIdx.x & 31, wv = threadIdx.x >> 5;
  __shared__ float sh[8];

  float mx = -3.0e38f;
  for (int t = threadIdx.x; t < S; t += 256) if (t <= s) mx = fmaxf(mx, row[t]);
  mx = wredMax(mx);
  if (lane == 0) sh[wv] = mx;
  __syncthreads();
  float m2 = sh[0];
  #pragma unroll
  for (int i = 1; i < 8; ++i) m2 = fmaxf(m2, sh[i]);
  __syncthreads();

  float sum = 0.f;
  for (int t = threadIdx.x; t < S; t += 256) {
    float e = (t <= s) ? __expf(row[t] - m2) : 0.0f;   // masked -> exactly 0
    row[t] = e; sum += e;
  }
  sum = wredSum(sum);
  if (lane == 0) sh[wv] = sum;
  __syncthreads();
  float s2 = 0.f;
  #pragma unroll
  for (int i = 0; i < 8; ++i) s2 += sh[i];
  float inv = 1.0f / s2;
  for (int t = threadIdx.x; t < S; t += 256) row[t] *= inv;
}

__global__ void k_zero_i32(int* p, int n) {
  int i = blockIdx.x * blockDim.x + threadIdx.x;
  if (i < n) p[i] = 0;
}

__global__ __launch_bounds__(256) void k_router(const float* __restrict__ gate,
                                                int* __restrict__ topi,
                                                float* __restrict__ topp,
                                                int* __restrict__ cnt) {
  int t = blockIdx.x * blockDim.x + threadIdx.x;
  if (t >= T_) return;
  float g[E_];
  #pragma unroll
  for (int e = 0; e < E_; ++e) g[e] = gate[t * E_ + e];
  int idx[TOPK_]; float val[TOPK_];
  #pragma unroll
  for (int j = 0; j < TOPK_; ++j) {
    int bi = 0; float bv = -3.4e38f;
    #pragma unroll
    for (int e = 0; e < E_; ++e) if (g[e] > bv) { bv = g[e]; bi = e; }
    idx[j] = bi; val[j] = bv; g[bi] = -3.4e38f;
  }
  // softmax over masked gate == softmax over top-k (exp(-1e20) == 0 in fp32)
  float mx = val[0], se = 0.f, ev[TOPK_];
  #pragma unroll
  for (int j = 0; j < TOPK_; ++j) { ev[j] = __expf(val[j] - mx); se += ev[j]; }
  #pragma unroll
  for (int j = 0; j < TOPK_; ++j) {
    topi[t * TOPK_ + j] = idx[j];
    topp[t * TOPK_ + j] = ev[j] / se;
    atomicAdd(&cnt[idx[j]], 1);
  }
}

__global__ void k_scan(const int* __restrict__ cnt, int* __restrict__ eoff,
                       int* __restrict__ fill) {
  if (blockIdx.x == 0 && threadIdx.x == 0) {
    int a = 0;
    for (int e = 0; e < E_; ++e) { eoff[e] = a; a += cnt[e]; fill[e] = 0; }
    eoff[E_] = a;
  }
}

__global__ __launch_bounds__(256) void k_fill(const int* __restrict__ topi,
                                              const float* __restrict__ topp,
                                              const int* __restrict__ eoff,
                                              int* __restrict__ fill,
                                              int* __restrict__ stok,
                                              float* __restrict__ sw) {
  int t = blockIdx.x * blockDim.x + threadIdx.x;
  if (t >= T_) return;
  #pragma unroll
  for (int j = 0; j < TOPK_; ++j) {
    int e = topi[t * TOPK_ + j];
    int pos = atomicAdd(&fill[e], 1);
    stok[eoff[e] + pos] = t;
    sw[eoff[e] + pos] = topp[t * TOPK_ + j];
  }
}

// act = silu(h1)*h2 for cols < Kv, exact 0 in pad cols [Kv, Kp)
__global__ __launch_bounds__(256) void k_silu_mul(float* __restrict__ h1,
                                                  const float* __restrict__ h2,
                                                  long long n, int Kp, int Kv) {
  long long i = (long long)blockIdx.x * blockDim.x + threadIdx.x;
  if (i >= n) return;
  int k = (int)(i % Kp);
  float r = 0.0f;
  if (k < Kv) {
    float a = h1[i], b = h2[i];
    r = a * (1.0f / (1.0f + __expf(-a))) * b;
  }
  h1[i] = r;
}

// dst[r][0:Kp] = { src[r][0:K], zeros }   (rows become 16B-aligned)
__global__ __launch_bounds__(256) void k_padcopy(const float* __restrict__ src,
                                                 float* __restrict__ dst,
                                                 long long tot, int K, int Kp) {
  long long i = (long long)blockIdx.x * blockDim.x + threadIdx.x;
  if (i >= tot) return;
  long long r = i / Kp; int k = (int)(i - r * Kp);
  dst[i] = (k < K) ? src[r * K + k] : 0.0f;
}

// ---------------- host-side helper ----------------
static void gemm(hipStream_t s,
                 const float* A, const float* W, float* C,
                 int M, int N, int K,
                 long long sAm, long long sAk, long long sWn, long long sWk,
                 long long ldc,
                 int nb1 = 1, int nb2 = 1,
                 long long bA1 = 0, long long bA2 = 0,
                 long long bW1 = 0, long long bW2 = 0,
                 long long bC1 = 0, long long bC2 = 0,
                 int flags = 0, float alpha = 1.0f, const float* aux = nullptr,
                 const int* expOff = nullptr, const int* slotTok = nullptr,
                 const float* slotW = nullptr) {
  GemmArgs p;
  p.A = A; p.W = W; p.C = C; p.aux = aux;
  p.expOff = expOff; p.slotTok = slotTok; p.slotW = slotW;
  p.sAm = sAm; p.sAk = sAk; p.sWn = sWn; p.sWk = sWk; p.ldc = ldc;
  p.bA1 = bA1; p.bA2 = bA2; p.bW1 = bW1; p.bW2 = bW2; p.bC1 = bC1; p.bC2 = bC2;
  p.nb2 = nb2; p.M = M; p.N = N; p.K = K; p.flags = flags; p.alpha = alpha;
  dim3 grid((N + 63) / 64, (M + 127) / 128, nb1 * nb2);
  bool fast = (sAk == 1) && ((sAm & 3) == 0) &&
              (sWk == 1) && ((sWn & 3) == 0) && ((K & 31) == 0);
  if (fast) k_gemm<true ><<<grid, dim3(128, 1, 1), 0, s>>>(p);
  else      k_gemm<false><<<grid, dim3(128, 1, 1), 0, s>>>(p);
}

extern "C" void kernel_launch(void* const* d_in, const int* in_sizes, int n_in,
                              void* d_out, int out_size, void* d_ws, size_t ws_size,
                              hipStream_t stream) {
  (void)in_sizes; (void)n_in; (void)out_size; (void)ws_size;
  const float* x     = (const float*)d_in[0];
  const float* g1    = (const float*)d_in[1];
  const float* g2    = (const float*)d_in[2];
  const float* Wdkv  = (const float*)d_in[3];   // [H,LAT,D] -> flat [512,512]
  const float* Wk    = (const float*)d_in[4];   // [H,HS,LAT]
  const float* Wv    = (const float*)d_in[5];   // [H,HS,LAT]
  const float* Wq    = (const float*)d_in[6];   // [H,HS,D]
  const float* Wo    = (const float*)d_in[7];   // [D,D]
  const float* Wg    = (const float*)d_in[8];   // [E,D]
  const float* rbias = (const float*)d_in[9];   // [E]
  const float* w1    = (const float*)d_in[10];  // [E,HID,D]
  const float* w2    = (const float*)d_in[11];  // [E,HID,D]
  const float* w3    = (const float*)d_in[12];  // [E,D,HID]
  const float* w1s   = (const float*)d_in[13];  // [HID,D]
  const float* w2s   = (const float*)d_in[14];  // [HID,D]
  const float* w3s   = (const float*)d_in[15];  // [D,HID]
  float* out = (float*)d_out;

  // ---- workspace bump allocator ----
  char* base = (char*)d_ws; size_t cur = 0;
  auto alloc = [&](size_t bytes) -> void* {
    void* p = base + cur; cur += (bytes + 255) & ~(size_t)255; return p;
  };
  float* xn     = (float*)alloc((size_t)T_ * D_ * 4);
  float* latall = (float*)alloc((size_t)T_ * 512 * 4);     // [T, H*LAT]
  float* absT   = (float*)alloc((size_t)512 * 512 * 4);    // [H*LAT, D]
  float* qres   = (float*)alloc((size_t)T_ * 512 * 4);
  float* scores = (float*)alloc((size_t)B_ * H_ * S_ * S_ * 4);
  float* vbuf   = (float*)alloc((size_t)B_ * H_ * S_ * HS_ * 4);
  float* hout   = (float*)alloc((size_t)T_ * 512 * 4);
  float* xn2    = (float*)alloc((size_t)T_ * D_ * 4);
  float* gate   = (float*)alloc((size_t)T_ * E_ * 4);
  int*   topi   = (int*)alloc((size_t)T_ * TOPK_ * 4);
  float* topp   = (float*)alloc((size_t)T_ * TOPK_ * 4);
  int*   cnt    = (int*)alloc(32 * 4);
  int*   eoff   = (int*)alloc(32 * 4);
  int*   fill   = (int*)alloc(32 * 4);
  int*   stok   = (int*)alloc((size_t)T_ * TOPK_ * 4);
  float* sw     = (float*)alloc((size_t)T_ * TOPK_ * 4);
  float* h1     = (float*)alloc((size_t)T_ * TOPK_ * HIDP_ * 4);
  float* h2     = (float*)alloc((size_t)T_ * TOPK_ * HIDP_ * 4);
  float* w3p    = (float*)alloc((size_t)E_ * D_ * HIDP_ * 4);   // padded w3
  float* w3sp   = (float*)alloc((size_t)D_ * HIDP_ * 4);        // padded w3s

  // padded copies of w3/w3s so the eo GEMMs stage with b128 loads
  {
    long long tot = (long long)E_ * D_ * HIDP_;
    k_padcopy<<<(int)((tot + 255) / 256), 256, 0, stream>>>(w3, w3p, tot, HID_, HIDP_);
    long long tot2 = (long long)D_ * HIDP_;
    k_padcopy<<<(int)((tot2 + 255) / 256), 256, 0, stream>>>(w3s, w3sp, tot2, HID_, HIDP_);
  }

  // ================= attention =================
  k_rmsnorm<<<T_, 256, 0, stream>>>(x, g1, xn, D_);

  // all-head latents: latall = xn @ WdkvAll^T   [T, 512]
  gemm(stream, xn, Wdkv, latall, T_, 512, 512, 512, 1, 512, 1, 512);

  // absorbedT[h][l][d] = sum_hs Wk[h][hs][l] * Wq[h][hs][d]  (batched over H)
  gemm(stream, Wk, Wq, absT, LAT_, D_, HS_,
       /*sAm*/1, /*sAk*/LAT_, /*sWn*/1, /*sWk*/D_, /*ldc*/D_,
       H_, 1, (long long)HS_ * LAT_, 0, (long long)HS_ * D_, 0,
       (long long)LAT_ * D_, 0);

  // qres = xn @ absorbedT^T (flat [512,D], K-contiguous)  [T, 512]
  gemm(stream, xn, absT, qres, T_, 512, 512, 512, 1, 512, 1, 512);

  // scores[b,h] = 0.125 * qres_h[b] @ lat0[b]^T   (only head-0 chunk survives mask)
  gemm(stream, qres, latall, scores, S_, S_, LAT_,
       512, 1, 512, 1, S_,
       B_, H_, (long long)S_ * 512, 64, (long long)S_ * 512, 0,
       (long long)H_ * S_ * S_, (long long)S_ * S_, 0, 0.125f);

  k_softmax_causal<<<B_ * H_ * S_, 256, 0, stream>>>(scores, S_);

  // v[b,h] = lat0[b] @ Wv[h]^T   [S, HS]
  gemm(stream, latall, Wv, vbuf, S_, HS_, LAT_,
       512, 1, LAT_, 1, HS_,
       B_, H_, (long long)S_ * 512, 0, 0, (long long)HS_ * LAT_,
       (long long)H_ * S_ * HS_, (long long)S_ * HS_);

  // hout[b,:,h*64:] = probs[b,h] @ v[b,h]
  gemm(stream, scores, vbuf, hout, S_, HS_, S_,
       S_, 1, 1, HS_, 512,
       B_, H_, (long long)H_ * S_ * S_, (long long)S_ * S_,
       (long long)H_ * S_ * HS_, (long long)S_ * HS_,
       (long long)S_ * 512, 64);

  // out = x + hout @ Wo^T   (residual fused)
  gemm(stream, hout, Wo, out, T_, D_, 512, 512, 1, 512, 1, D_,
       1, 1, 0, 0, 0, 0, 0, 0, F_ADD, 1.0f, x);

  // ================= MoE =================
  k_rmsnorm<<<T_, 256, 0, stream>>>(out, g2, xn2, D_);

  // gate = xn2 @ Wg^T + bias
  gemm(stream, xn2, Wg, gate, T_, E_, 512, 512, 1, 512, 1, E_,
       1, 1, 0, 0, 0, 0, 0, 0, F_BIAS, 1.0f, rbias);

  k_zero_i32<<<1, 64, 0, stream>>>(cnt, E_);
  k_router<<<T_ / 256, 256, 0, stream>>>(gate, topi, topp, cnt);
  k_scan<<<1, 1, 0, stream>>>(cnt, eoff, fill);
  k_fill<<<T_ / 256, 256, 0, stream>>>(topi, topp, eoff, fill, stok, sw);

  // routed experts: h1/h2 (token-gathered), only top-k (exact: softmax zeros)
  gemm(stream, xn2, w1, h1, T_, HID_, D_, 512, 1, 512, 1, HIDP_,
       E_, 1, 0, 0, (long long)HID_ * D_, 0, 0, 0,
       F_EXPERT | F_AGATHER, 1.0f, nullptr, eoff, stok, nullptr);
  gemm(stream, xn2, w2, h2, T_, HID_, D_, 512, 1, 512, 1, HIDP_,
       E_, 1, 0, 0, (long long)HID_ * D_, 0, 0, 0,
       F_EXPERT | F_AGATHER, 1.0f, nullptr, eoff, stok, nullptr);
  k_silu_mul<<<(int)(((long long)T_ * TOPK_ * HIDP_ + 255) / 256), 256, 0, stream>>>(
      h1, h2, (long long)T_ * TOPK_ * HIDP_, HIDP_, HID_);

  // eo: out[token] += prob * (act @ w3p[e]^T)  (K padded: pad cols are exact 0)
  gemm(stream, h1, w3p, out, T_, D_, HIDP_, HIDP_, 1, HIDP_, 1, D_,
       E_, 1, 0, 0, (long long)D_ * HIDP_, 0, 0, 0,
       F_EXPERT | F_SCATTER, 1.0f, nullptr, eoff, stok, sw);

  // shared expert (reuse h2 and scores buffers, both large enough)
  gemm(stream, xn2, w1s, h2, T_, HID_, D_, 512, 1, 512, 1, HIDP_);
  gemm(stream, xn2, w2s, scores, T_, HID_, D_, 512, 1, 512, 1, HIDP_);
  k_silu_mul<<<(int)(((long long)T_ * HIDP_ + 255) / 256), 256, 0, stream>>>(
      h2, scores, (long long)T_ * HIDP_, HIDP_, HID_);
  gemm(stream, h2, w3sp, out, T_, D_, HIDP_, HIDP_, 1, HIDP_, 1, D_,
       1, 1, 0, 0, 0, 0, 0, 0, F_SCATTER, 1.0f, nullptr, nullptr, nullptr, nullptr);
}
